// VimBlock_51891794871099
// MI455X (gfx1250) — matrix operand order repaired and checked by reading, not compiled, not run on hardware
//
#include <hip/hip_runtime.h>
#include <cstdint>
#include <cstddef>

// ---------------------------------------------------------------------------
// Vim (bidirectional Mamba) block for MI455X / gfx1250, wave32 + WMMA bf16.
// B=4, L=2048, D=768, DI=1536, DS=16, DTR=48, DCONV=4
// ---------------------------------------------------------------------------

#define B_ 4
#define L_ 2048
#define D_ 768
#define DI_ 1536
#define DS_ 16
#define DTR_ 48
#define DCONV_ 4
#define ROWS_ (B_ * L_)        // 8192
#define ROWS2_ (2 * ROWS_)     // 16384 (both directions)

typedef __attribute__((ext_vector_type(16))) __bf16 v16bf;
typedef __attribute__((ext_vector_type(8)))  float  v8f;
typedef __attribute__((ext_vector_type(4)))  float  f4x;
typedef __attribute__((__vector_size__(4 * sizeof(int)))) int i32x4;

#define GLOBAL_AS __attribute__((address_space(1)))
#define LDS_AS    __attribute__((address_space(3)))

// gfx1250 async global->LDS copy (ASYNCcnt path), with portable fallback.
#if defined(__has_builtin)
#if __has_builtin(__builtin_amdgcn_global_load_async_to_lds_b128)
#define USE_ASYNC_LDS 1
#endif
#endif

// ---------------------------------------------------------------------------
// Weight conversion: src (K x N) f32 row-major  ->  dst (Nrows x Kpad) bf16
// (i.e. B^T, contiguous along the reduction dim; zero-padded)
// ---------------------------------------------------------------------------
__global__ void k_conv_trans(const float* __restrict__ src, __bf16* __restrict__ dst,
                             int K, int N, int Kpad, int Nrows) {
    size_t total = (size_t)Nrows * Kpad;
    for (size_t i = (size_t)blockIdx.x * blockDim.x + threadIdx.x; i < total;
         i += (size_t)gridDim.x * blockDim.x) {
        int n = (int)(i / Kpad);
        int k = (int)(i % Kpad);
        float v = (k < K && n < N) ? src[(size_t)k * N + n] : 0.0f;
        dst[i] = (__bf16)v;
    }
}

// W_out (1536 x 768) -> transposed (768 x 3072): dst[n,k] = W[(k & 1535), n]
__global__ void k_build_woutT2(const float* __restrict__ W, __bf16* __restrict__ dst) {
    size_t total = (size_t)D_ * (2 * DI_);
    for (size_t i = (size_t)blockIdx.x * blockDim.x + threadIdx.x; i < total;
         i += (size_t)gridDim.x * blockDim.x) {
        int n = (int)(i / (2 * DI_));
        int k = (int)(i % (2 * DI_));
        dst[i] = (__bf16)W[(size_t)(k & (DI_ - 1)) * D_ + n];
    }
}

// ---------------------------------------------------------------------------
// LayerNorm over D=768, one block per row, emits bf16 A-operand for the GEMM.
// ---------------------------------------------------------------------------
__global__ void __launch_bounds__(256) k_layernorm(const float* __restrict__ x,
                                                   const float* __restrict__ w,
                                                   const float* __restrict__ bias,
                                                   __bf16* __restrict__ xn) {
    const int row = blockIdx.x;
    const int tid = threadIdx.x;
    const float* xr = x + (size_t)row * D_;
    float v[3];
    float s = 0.f, s2 = 0.f;
#pragma unroll
    for (int i = 0; i < 3; ++i) {
        v[i] = xr[tid + i * 256];
        s += v[i];
        s2 += v[i] * v[i];
    }
    __shared__ float rs[256];
    __shared__ float rq[256];
    rs[tid] = s; rq[tid] = s2;
    __syncthreads();
    for (int off = 128; off > 0; off >>= 1) {
        if (tid < off) { rs[tid] += rs[tid + off]; rq[tid] += rq[tid + off]; }
        __syncthreads();
    }
    const float mu  = rs[0] * (1.0f / D_);
    const float var = rq[0] * (1.0f / D_) - mu * mu;
    const float rinv = rsqrtf(var + 1e-5f);
    __bf16* outr = xn + (size_t)row * D_;
#pragma unroll
    for (int i = 0; i < 3; ++i) {
        int c = tid + i * 256;
        outr[c] = (__bf16)((v[i] - mu) * rinv * w[c] + bias[c]);
    }
}

// ---------------------------------------------------------------------------
// bf16 WMMA GEMM with LDS double-buffered staging.
//   C(MxN,f32) = A(MxK,bf16 rowmajor) * B,  Bt = B^T stored (Npad x K) bf16.
// Block tile 64x256 (8 waves, 2M x 4N), wave tile 32x64 (2 A frags x 4 B
// frags -> 8 WMMAs per chunk), K chunk = 32. Per chunk: sB (256x32, 16KB) +
// sA (64x32, 4KB) staged via async global->LDS (ASYNCcnt) when available,
// else register + ds_store. Copy of chunk i+1 overlaps compute of chunk i.
// Each B fragment feeds two back-to-back WMMAs. K%32==0, M%64==0. Epilogues:
//   0: plain store   1: softplus(acc + bias[col])   2: resid + scale*acc
// ---------------------------------------------------------------------------
union FragU { f4x f4[2]; v16bf v; };

__global__ void __launch_bounds__(256) k_gemm_bf16(
    const __bf16* __restrict__ A, const __bf16* __restrict__ Bt,
    float* __restrict__ C, int M, int N, int K,
    int epi, const float* __restrict__ bias,
    const float* __restrict__ resid, float scale) {
    __shared__ __bf16 sA[2][64 * 32];
    __shared__ __bf16 sB[2][256 * 32];

    const int tid    = threadIdx.x;
    const int wave   = tid >> 5;
    const int lane   = tid & 31;
    const int laneLo = lane & 15;
    const int laneHi = lane >> 4;          // 0 or 1 (wave32 halves)
    const int waveM  = wave & 1;
    const int waveN  = wave >> 1;
    const int rowBlk = blockIdx.y * 64;
    const int colBlk = blockIdx.x * 256;
    const int row0 = rowBlk + waveM * 32;
    const int col0 = colBlk + waveN * 64;

    // Staging map (all 256 threads):
    //   B tile 256 rows x 32 k : 4 x b128 per thread
    //   A tile  64 rows x 32 k : 1 x b128 per thread (64 rows x 4 segs)
    const int aRow = tid >> 2;
    const int aSeg = tid & 3;
    const __bf16* gA = A + (size_t)(rowBlk + aRow) * K + aSeg * 8;

    // compute-side LDS offsets (elements)
    const int ka = laneHi * 8;             // ISA 16-bit A fragment K split
    const int kb = laneHi * 8;             // ISA 16-bit B fragment K split
    int aFragOff[2];
#pragma unroll
    for (int m = 0; m < 2; ++m)
        aFragOff[m] = (waveM * 32 + m * 16 + laneLo) * 32 + ka;
    int bFragOff[4];
#pragma unroll
    for (int j = 0; j < 4; ++j)
        bFragOff[j] = (waveN * 64 + j * 16 + laneLo) * 32 + kb;

#if !defined(USE_ASYNC_LDS)
    f4x rB[4];
    f4x rA;
#endif

    v8f acc[8] = {};   // [m*4 + j]

    const int nk = K >> 5;

    // ---- staging helpers ----
    auto stage_issue = [&](int buf, int k) {
#pragma unroll
        for (int p = 0; p < 4; ++p) {
            const int idx = p * 256 + tid;
            const int row = idx >> 2;
            const int seg = idx & 3;
            const __bf16* g = Bt + (size_t)(colBlk + row) * K + k + seg * 8;
            __bf16* l = &sB[buf][row * 32 + seg * 8];
#if defined(USE_ASYNC_LDS)
            __builtin_amdgcn_global_load_async_to_lds_b128(
                (GLOBAL_AS i32x4*)g, (LDS_AS i32x4*)l, 0, 0);
#else
            rB[p] = *(const f4x*)g;
            (void)l;
#endif
        }
        {
            const __bf16* g = gA + k;
            __bf16* l = &sA[buf][aRow * 32 + aSeg * 8];
#if defined(USE_ASYNC_LDS)
            __builtin_amdgcn_global_load_async_to_lds_b128(
                (GLOBAL_AS i32x4*)g, (LDS_AS i32x4*)l, 0, 0);
#else
            rA = *(const f4x*)g;
            (void)l;
#endif
        }
    };
    auto stage_commit = [&](int buf) {
#if defined(USE_ASYNC_LDS)
        asm volatile("s_wait_asynccnt 0x0" ::: "memory");
        (void)buf;
#else
#pragma unroll
        for (int p = 0; p < 4; ++p) {
            const int idx = p * 256 + tid;
            const int row = idx >> 2;
            const int seg = idx & 3;
            *(f4x*)&sB[buf][row * 32 + seg * 8] = rB[p];
        }
        *(f4x*)&sA[buf][aRow * 32 + aSeg * 8] = rA;
#endif
    };

    // ---- pipeline ----
    stage_issue(0, 0);
    stage_commit(0);
    __syncthreads();

    for (int i = 0; i < nk; ++i) {
        const int cur = i & 1;
        const int nxt = cur ^ 1;
        if (i + 1 < nk) stage_issue(nxt, (i + 1) << 5);

        FragU a0, a1;
        a0.f4[0] = *(const f4x*)&sA[cur][aFragOff[0]];
        a0.f4[1] = *(const f4x*)&sA[cur][aFragOff[0] + 16];
        a1.f4[0] = *(const f4x*)&sA[cur][aFragOff[1]];
        a1.f4[1] = *(const f4x*)&sA[cur][aFragOff[1] + 16];
#pragma unroll
        for (int j = 0; j < 4; ++j) {
            FragU b;
            b.f4[0] = *(const f4x*)&sB[cur][bFragOff[j]];
            b.f4[1] = *(const f4x*)&sB[cur][bFragOff[j] + 16];
            acc[j] = __builtin_amdgcn_wmma_f32_16x16x32_bf16(
                false, a0.v, false, b.v, (short)0, acc[j], false, false);
            acc[4 + j] = __builtin_amdgcn_wmma_f32_16x16x32_bf16(
                false, a1.v, false, b.v, (short)0, acc[4 + j], false, false);
        }

        if (i + 1 < nk) stage_commit(nxt);
        __syncthreads();
    }

    // D layout: VGPR r -> row = row0 + m*16 + r + 8*laneHi, col = col0 + 16j + laneLo
#pragma unroll
    for (int m = 0; m < 2; ++m) {
#pragma unroll
        for (int j = 0; j < 4; ++j) {
            const int col = col0 + j * 16 + laneLo;
            if (col < N) {
#pragma unroll
                for (int r = 0; r < 8; ++r) {
                    const int row = row0 + m * 16 + laneHi * 8 + r;
                    float val = acc[m * 4 + j][r];
                    const size_t o = (size_t)row * N + col;
                    if (epi == 1) {
                        val += bias[col];
                        val = (val > 20.0f) ? val : log1pf(__expf(val));
                        C[o] = val;
                    } else if (epi == 2) {
                        C[o] = resid[o] + scale * val;
                    } else {
                        C[o] = val;
                    }
                }
            }
        }
    }
}

// ---------------------------------------------------------------------------
// Causal depthwise conv (K=4) + SiLU for both directions.
// fwd: taps l-3..l  (w[0]..w[3]);  bwd (flipped time): taps l+3..l.
// ---------------------------------------------------------------------------
__global__ void k_conv_silu(const float* __restrict__ xz,
                            const float* __restrict__ cw,
                            const float* __restrict__ cb,
                            __bf16* __restrict__ u) {
    const size_t total = (size_t)2 * B_ * L_ * DI_;
    for (size_t i = (size_t)blockIdx.x * blockDim.x + threadIdx.x; i < total;
         i += (size_t)gridDim.x * blockDim.x) {
        const int c   = (int)(i % DI_);
        const int l   = (int)((i / DI_) % L_);
        const int b   = (int)((i / ((size_t)DI_ * L_)) % B_);
        const int dir = (int)(i / ((size_t)DI_ * L_ * B_));
        float acc = cb[c];
#pragma unroll
        for (int k = 0; k < DCONV_; ++k) {
            const int ll = dir ? (l + 3 - k) : (l - 3 + k);
            if (ll >= 0 && ll < L_)
                acc += xz[((size_t)(b * L_ + ll)) * (2 * DI_) + c] * cw[c * DCONV_ + k];
        }
        const float sv = acc / (1.0f + __expf(-acc));   // silu
        u[((size_t)((dir * B_ + b) * L_ + l)) * DI_ + c] = (__bf16)sv;
    }
}

// dt_raw (cols 0..47 of xdb, f32) -> bf16 padded to K=64
__global__ void k_pad_dtraw(const float* __restrict__ xdb, __bf16* __restrict__ dst) {
    const size_t total = (size_t)ROWS2_ * 64;
    for (size_t i = (size_t)blockIdx.x * blockDim.x + threadIdx.x; i < total;
         i += (size_t)gridDim.x * blockDim.x) {
        const size_t r = i >> 6;
        const int k = (int)(i & 63);
        dst[i] = (k < DTR_) ? (__bf16)xdb[r * 80 + k] : (__bf16)0.0f;
    }
}

// ---------------------------------------------------------------------------
// Selective scan. One lane per (dir,b,di) channel; 16-wide state in VGPRs.
// Fused epilogue: yact = (y + u*D) * silu(z), stored bf16 at concat-K layout.
// ---------------------------------------------------------------------------
union Ld16 { f4x q[4]; float f[16]; };

__global__ void __launch_bounds__(256) k_scan(
    const float* __restrict__ dt, const float* __restrict__ xdb,
    const __bf16* __restrict__ u, const float* __restrict__ xz,
    const float* __restrict__ A_log, const float* __restrict__ Dp,
    __bf16* __restrict__ yact) {
    const int gwave = (int)((blockIdx.x * blockDim.x + threadIdx.x) >> 5);
    const int lane  = threadIdx.x & 31;
    // 2*B*(DI/32) = 384 waves
    const int grp = gwave % (DI_ / 32);
    const int b   = (gwave / (DI_ / 32)) % B_;
    const int dir = gwave / (B_ * (DI_ / 32));
    const int di  = grp * 32 + lane;

    float Arow[DS_];
#pragma unroll
    for (int s = 0; s < DS_; ++s) Arow[s] = -__expf(A_log[di * DS_ + s]);
    const float dparam = Dp[di];

    float h[DS_];
#pragma unroll
    for (int s = 0; s < DS_; ++s) h[s] = 0.0f;

    for (int t = 0; t < L_; ++t) {
        const int l = dir ? (L_ - 1 - t) : t;
        const size_t row = (size_t)(dir * B_ + b) * L_ + l;
        const float dtv = dt[row * DI_ + di];
        const float uv  = (float)u[row * DI_ + di];
        const float* bc = xdb + row * 80 + DTR_;
        Ld16 Bv, Cv;
#pragma unroll
        for (int q = 0; q < 4; ++q) Bv.q[q] = *(const f4x*)(bc + 4 * q);
#pragma unroll
        for (int q = 0; q < 4; ++q) Cv.q[q] = *(const f4x*)(bc + DS_ + 4 * q);
        const float du = dtv * uv;
        float y = 0.0f;
#pragma unroll
        for (int s = 0; s < DS_; ++s) {
            h[s] = h[s] * __expf(dtv * Arow[s]) + du * Bv.f[s];
            y += h[s] * Cv.f[s];
        }
        const size_t orow = (size_t)(b * L_ + l);
        const float z = xz[orow * (2 * DI_) + DI_ + di];
        const float sz = z / (1.0f + __expf(-z));
        const float out = (y + uv * dparam) * sz;
        yact[orow * (2 * DI_) + dir * DI_ + di] = (__bf16)out;
    }
}

// ---------------------------------------------------------------------------
// Host launcher
// ---------------------------------------------------------------------------
extern "C" void kernel_launch(void* const* d_in, const int* in_sizes, int n_in,
                              void* d_out, int out_size, void* d_ws, size_t ws_size,
                              hipStream_t stream) {
    const float* x      = (const float*)d_in[0];
    const float* ln_w   = (const float*)d_in[1];
    const float* ln_b   = (const float*)d_in[2];
    const float* W_in   = (const float*)d_in[3];
    const float* conv_w = (const float*)d_in[4];
    const float* conv_b = (const float*)d_in[5];
    const float* W_x    = (const float*)d_in[6];
    const float* W_dt   = (const float*)d_in[7];
    const float* b_dt   = (const float*)d_in[8];
    const float* A_log  = (const float*)d_in[9];
    const float* D_par  = (const float*)d_in[10];
    const float* W_out  = (const float*)d_in[11];
    float* out = (float*)d_out;

    // -------- workspace carve-up --------
    char* base = (char*)d_ws;
    size_t off = 0;
    auto carve = [&](size_t bytes) -> char* {
        char* p = base + off;
        off = (off + bytes + 255) & ~(size_t)255;
        return p;
    };
    __bf16* xn      = (__bf16*)carve((size_t)ROWS_ * D_ * 2);              // 12.6 MB
    __bf16* W_inT   = (__bf16*)carve((size_t)(2 * DI_) * D_ * 2);          //  4.7 MB (3072x768)
    __bf16* W_xT    = (__bf16*)carve((size_t)256 * DI_ * 2);               //  0.8 MB (pad N 80->256)
    __bf16* W_dtT   = (__bf16*)carve((size_t)DI_ * 64 * 2);                //  0.2 MB (pad K 48->64)
    __bf16* W_outT2 = (__bf16*)carve((size_t)D_ * (2 * DI_) * 2);          //  4.7 MB (768x3072)
    float*  xz      = (float*)carve((size_t)ROWS_ * (2 * DI_) * 4);        // 100 MB
    __bf16* u_dir   = (__bf16*)carve((size_t)ROWS2_ * DI_ * 2);            //  50 MB
    float*  xdb     = (float*)carve((size_t)ROWS2_ * 80 * 4);              //  5.2 MB
    __bf16* dtraw   = (__bf16*)carve((size_t)ROWS2_ * 64 * 2);             //  2.1 MB
    float*  dtbuf   = (float*)carve((size_t)ROWS2_ * DI_ * 4);             // 100 MB
    __bf16* yact    = (__bf16*)carve((size_t)ROWS_ * (2 * DI_) * 2);       //  50 MB
    (void)ws_size; (void)n_in; (void)in_sizes; (void)out_size;

    // -------- weight preprocessing (bf16, transposed for contiguous-K frags) --------
    {
        int tb = 256;
        k_conv_trans<<<dim3(4608), tb, 0, stream>>>(W_in, W_inT, D_, 2 * DI_, D_, 2 * DI_);
        k_conv_trans<<<dim3(1536), tb, 0, stream>>>(W_x,  W_xT,  DI_, 80, DI_, 256);
        k_conv_trans<<<dim3(384),  tb, 0, stream>>>(W_dt, W_dtT, DTR_, DI_, 64, DI_);
        k_build_woutT2<<<dim3(4608), tb, 0, stream>>>(W_out, W_outT2);
    }

    // -------- 1. LayerNorm -> bf16 --------
    k_layernorm<<<dim3(ROWS_), 256, 0, stream>>>(x, ln_w, ln_b, xn);

    // -------- 2. xz = xn @ W_in : (8192 x 3072 x K=768) --------
    k_gemm_bf16<<<dim3((2 * DI_) / 256, ROWS_ / 64), 256, 0, stream>>>(
        xn, W_inT, xz, ROWS_, 2 * DI_, D_, 0, nullptr, nullptr, 0.f);

    // -------- 3. depthwise conv + SiLU (both directions) --------
    k_conv_silu<<<dim3(98304), 256, 0, stream>>>(xz, conv_w, conv_b, u_dir);

    // -------- 4. xdb = u @ W_x : (16384 x 80 x K=1536) --------
    k_gemm_bf16<<<dim3(1, ROWS2_ / 64), 256, 0, stream>>>(
        u_dir, W_xT, xdb, ROWS2_, 80, DI_, 0, nullptr, nullptr, 0.f);

    // -------- 5. dt = softplus(dt_raw @ W_dt + b_dt) : (16384 x 1536 x K=64) --------
    k_pad_dtraw<<<dim3(4096), 256, 0, stream>>>(xdb, dtraw);
    k_gemm_bf16<<<dim3(DI_ / 256, ROWS2_ / 64), 256, 0, stream>>>(
        dtraw, W_dtT, dtbuf, ROWS2_, DI_, 64, 1, b_dt, nullptr, 0.f);

    // -------- 6. selective scan (both directions) + fused gating --------
    k_scan<<<dim3(48), 256, 0, stream>>>(dtbuf, xdb, u_dir, xz, A_log, D_par, yact);

    // -------- 7. out = x + 0.5*([yf|yb] @ W_out[k & 1535, :]) : (8192 x 768 x K=3072) --------
    k_gemm_bf16<<<dim3(D_ / 256, ROWS_ / 64), 256, 0, stream>>>(
        yact, W_outT2, out, ROWS_, D_, 2 * DI_, 2, nullptr, x, 0.5f);
}
